// GCN_39453569581377
// MI455X (gfx1250) — compile-verified, matrix operand order checked by reading, not run on hardware
//
#include <hip/hip_runtime.h>
#include <hip/hip_bf16.h>

// ---------------- problem constants (match reference) ----------------
#define NNODES 50000
#define NEDGES 800000
#define DIN    64
#define DH     128          // hidden dim
#define NG     256          // graphs
#define NSTEPS 4
#define NCONV  3

#define TILES_M   (NNODES / 16)   // 3125
#define GEMM_TPB  16              // row tiles per block (8 per wave-group)
#define GRU_TPB   16

typedef __attribute__((ext_vector_type(16))) __bf16 v16bf;
typedef __attribute__((ext_vector_type(8)))  __bf16 v8bf;
typedef __attribute__((ext_vector_type(8)))  float  v8f;

// ---------------- WMMA helpers (gfx1250, wave32) ----------------

__device__ __forceinline__ v8f wmma_bf16(v16bf a, v16bf b, v8f c) {
  // D = A(16x32 bf16) x B(32x16 bf16) + C(16x16 f32)
  return __builtin_amdgcn_wmma_f32_16x16x32_bf16(
      /*neg_a=*/false, a, /*neg_b=*/false, b,
      /*c_mod=*/(short)0, c, /*reuse_a=*/false, /*reuse_b=*/false);
}

// 16x32 bf16 fragment load (A layout; B uses same striping with lane = N,
// W row-major [out][in] so B = W^T reads are two contiguous 16B chunks).
__device__ __forceinline__ v16bf ld_frag(const __bf16* __restrict__ base,
                                         int ld, int rc0, int k0) {
  const int lane = threadIdx.x & 31;
  const int grp  = lane >> 4;
  const int r    = lane & 15;
  const __bf16* p = base + (size_t)(rc0 + r) * (size_t)ld + k0 + 8 * grp;
  v8bf lo = *(const v8bf*)p;
  v8bf hi = *(const v8bf*)(p + 16);
  v16bf f;
#pragma unroll
  for (int i = 0; i < 8; ++i) { f[i] = lo[i]; f[i + 8] = hi[i]; }
  return f;
}

// LDS flavor (pointer derived from __shared__; lowers to ds_load_b128)
__device__ __forceinline__ v16bf ld_frag_lds(const __bf16* base,
                                             int ld, int rc0, int k0) {
  const int lane = threadIdx.x & 31;
  const int grp  = lane >> 4;
  const int r    = lane & 15;
  const __bf16* p = base + (rc0 + r) * ld + k0 + 8 * grp;
  v8bf lo = *(const v8bf*)p;
  v8bf hi = *(const v8bf*)(p + 16);
  v16bf f;
#pragma unroll
  for (int i = 0; i < 8; ++i) { f[i] = lo[i]; f[i + 8] = hi[i]; }
  return f;
}

// async stage `bytes16` 16B-chunks from global -> LDS (uniform loop, EXEC full)
__device__ __forceinline__ void stage_async(unsigned lds_base,
                                            const __bf16* gsrc, int chunks) {
  unsigned long long g = (unsigned long long)(uintptr_t)gsrc;
  for (int i = threadIdx.x; i < chunks; i += (int)blockDim.x) {
    unsigned      l = lds_base + i * 16;
    unsigned long long a = g + (unsigned long long)i * 16;
    asm volatile("global_load_async_to_lds_b128 %0, %1, off"
                 :: "v"(l), "v"(a) : "memory");
  }
}

__device__ __forceinline__ void wait_async_all() {
  asm volatile("s_wait_asynccnt 0x0" ::: "memory");
}

// ---------------- generic GEMM:  C[M,Nout] = A[M,K] (bf16) @ W[Nout,K]^T ----
// 256 threads = 2 wave-groups x 4 waves; weight panel (128 cols x K) staged in
// LDS via async loads; each block covers GEMM_TPB row-tiles.
__global__ __launch_bounds__(256)
void ggc_gemm_bf16(const __bf16* __restrict__ A, const __bf16* __restrict__ W,
                   const float* __restrict__ bias,
                   float* __restrict__ C, __bf16* __restrict__ Cb,
                   int nTileM, int K, int Nout, int relu) {
  extern __shared__ __bf16 sW[];   // [128][K]
  // ---- stage weight panel for this blockIdx.y ----
  stage_async((unsigned)(uintptr_t)sW,
              W + (size_t)blockIdx.y * 128 * K, 128 * K * 2 / 16);
  wait_async_all();
  __syncthreads();

  const int cwave = (threadIdx.x >> 5) & 3;   // column group (32 cols)
  const int wg    = threadIdx.x >> 7;         // wave-group 0/1
  const int lane  = threadIdx.x & 31, grp = lane >> 4, n = lane & 15;

  const int tCap = blockIdx.x * GEMM_TPB + GEMM_TPB;
  const int tEnd = tCap < nTileM ? tCap : nTileM;
  for (int t0 = blockIdx.x * GEMM_TPB + wg; t0 < tEnd; t0 += 2) {
    const int row0 = t0 * 16;
    v8f acc0 = {};
    v8f acc1 = {};
    for (int k0 = 0; k0 < K; k0 += 32) {
      v16bf a = ld_frag(A, K, row0, k0);
      if (k0 + 32 < K)
        __builtin_prefetch((const void*)(A + (size_t)(row0 + n) * K + k0 + 32), 0, 1);
      v16bf b0 = ld_frag_lds(sW, K, cwave * 32,      k0);
      v16bf b1 = ld_frag_lds(sW, K, cwave * 32 + 16, k0);
      acc0 = wmma_bf16(a, b0, acc0);
      acc1 = wmma_bf16(a, b1, acc1);
    }
#pragma unroll
    for (int t = 0; t < 2; ++t) {
      v8f acc = t ? acc1 : acc0;
      const int col = blockIdx.y * 128 + cwave * 32 + t * 16 + n;
      const float bv = bias ? bias[col] : 0.f;
#pragma unroll
      for (int e = 0; e < 8; ++e) {
        const int row = row0 + e + 8 * grp;
        float v = acc[e] + bv;
        if (relu) v = fmaxf(v, 0.f);
        C[(size_t)row * Nout + col] = v;
        if (Cb) Cb[(size_t)row * Nout + col] = (__bf16)v;
      }
    }
  }
}

// ---------------- fused GRU cell (ping-pong h, LDS-resident weights) --------
// gi = agg @ wih^T + bih ; gh = h @ whh^T + bhh   (wih/whh: [384,128])
// r = sig(i_r+h_r); z = sig(i_z+h_z); n = tanh(i_n + r*h_n)
// h' = (1-z)*n + z*h  (optional relu).  Reads hin/hBin, writes hout/hBout.
__global__ __launch_bounds__(256)
void ggc_gru_fused(const __bf16* __restrict__ aggB, const __bf16* __restrict__ hBin,
                   const __bf16* __restrict__ wihB, const __bf16* __restrict__ whhB,
                   const float* __restrict__ bih, const float* __restrict__ bhh,
                   const float* __restrict__ hin, float* __restrict__ hout,
                   __bf16* __restrict__ hBout, int nTiles, int relu) {
  extern __shared__ __bf16 smem[];          // 2 x [384][128] = 192 KB
  __bf16* sWih = smem;
  __bf16* sWhh = smem + 3 * DH * DH;

  stage_async((unsigned)(uintptr_t)sWih, wihB, 3 * DH * DH * 2 / 16);
  stage_async((unsigned)(uintptr_t)sWhh, whhB, 3 * DH * DH * 2 / 16);
  wait_async_all();
  __syncthreads();

  const int cwave = (threadIdx.x >> 5) & 3;
  const int wg    = threadIdx.x >> 7;
  const int lane  = threadIdx.x & 31, grp = lane >> 4, n = lane & 15;

  const int tCap = blockIdx.x * GRU_TPB + GRU_TPB;
  const int tEnd = tCap < nTiles ? tCap : nTiles;
  for (int t0 = blockIdx.x * GRU_TPB + wg; t0 < tEnd; t0 += 2) {
    const int row0 = t0 * 16;
    v8f accI[6] = {};   // [gate(3)][ntile(2)] for this wave's 32 cols
    v8f accH[6] = {};
    for (int k0 = 0; k0 < DH; k0 += 32) {
      v16bf aA = ld_frag(aggB, DH, row0, k0);
      v16bf aH = ld_frag(hBin, DH, row0, k0);
#pragma unroll
      for (int c = 0; c < 3; ++c) {
#pragma unroll
        for (int t = 0; t < 2; ++t) {
          const int colW = c * DH + cwave * 32 + t * 16;
          v16bf bI = ld_frag_lds(sWih, DH, colW, k0);
          accI[c * 2 + t] = wmma_bf16(aA, bI, accI[c * 2 + t]);
          v16bf bH = ld_frag_lds(sWhh, DH, colW, k0);
          accH[c * 2 + t] = wmma_bf16(aH, bH, accH[c * 2 + t]);
        }
      }
    }
#pragma unroll
    for (int t = 0; t < 2; ++t) {
      const int col = cwave * 32 + t * 16 + n;
      const float b_ir = bih[col],          b_hr = bhh[col];
      const float b_iz = bih[DH + col],     b_hz = bhh[DH + col];
      const float b_in = bih[2 * DH + col], b_hn = bhh[2 * DH + col];
      v8f ir = accI[t], iz = accI[2 + t], inn = accI[4 + t];
      v8f hr = accH[t], hz = accH[2 + t], hnn = accH[4 + t];
#pragma unroll
      for (int e = 0; e < 8; ++e) {
        const size_t idx = (size_t)(row0 + e + 8 * grp) * DH + col;
        const float hold = hin[idx];
        const float r  = 1.f / (1.f + expf(-((ir[e] + b_ir) + (hr[e] + b_hr))));
        const float z  = 1.f / (1.f + expf(-((iz[e] + b_iz) + (hz[e] + b_hz))));
        const float nn = tanhf((inn[e] + b_in) + r * (hnn[e] + b_hn));
        float hv = (1.f - z) * nn + z * hold;
        if (relu) hv = fmaxf(hv, 0.f);
        hout[idx]  = hv;
        hBout[idx] = (__bf16)hv;
      }
    }
  }
}

// ---------------- edge scatter-add: agg[dst] += m[src] (L2-resident) --------
__global__ __launch_bounds__(256)
void ggc_edge_scatter(const float* __restrict__ m, const int* __restrict__ src,
                      const int* __restrict__ dst, float* __restrict__ agg, int E) {
  const int e = blockIdx.x * 8 + (threadIdx.x >> 5);
  if (e >= E) return;
  const int lane = threadIdx.x & 31;
  const int s = src[e], d = dst[e];
  const float4 v = ((const float4*)(m + (size_t)s * DH))[lane];
  float* a = agg + (size_t)d * DH + lane * 4;
  atomicAdd(a + 0, v.x); atomicAdd(a + 1, v.y);
  atomicAdd(a + 2, v.z); atomicAdd(a + 3, v.w);
}

// ---------------- global add pool into pooled[:, colOff:colOff+128] --------
__global__ __launch_bounds__(256)
void ggc_pool(const float* __restrict__ h, const int* __restrict__ batch,
              float* __restrict__ pooled, int N, int colOff) {
  const int node = blockIdx.x * 8 + (threadIdx.x >> 5);
  if (node >= N) return;
  const int lane = threadIdx.x & 31;
  const int g = batch[node];
  const float4 v = ((const float4*)(h + (size_t)node * DH))[lane];
  float* p = pooled + (size_t)g * (3 * DH) + colOff + lane * 4;
  atomicAdd(p + 0, v.x); atomicAdd(p + 1, v.y);
  atomicAdd(p + 2, v.z); atomicAdd(p + 3, v.w);
}

// ---------------- small utility kernels ----------------
__global__ void ggc_zero_f32(float* p, int n) {
  int i = blockIdx.x * blockDim.x + threadIdx.x;
  if (i < n) p[i] = 0.f;
}

__global__ void ggc_f32_to_bf16(const float* __restrict__ s, __bf16* __restrict__ d, int n) {
  int i = blockIdx.x * blockDim.x + threadIdx.x;
  if (i < n) d[i] = (__bf16)s[i];
}

__global__ void ggc_pad_x(const float* __restrict__ x, float* __restrict__ h,
                          __bf16* __restrict__ hB, int N) {
  int i = blockIdx.x * blockDim.x + threadIdx.x;
  if (i >= N * DH) return;
  const int node = i >> 7, c = i & 127;
  const float v = (c < DIN) ? x[(size_t)node * DIN + c] : 0.f;
  h[i]  = v;
  hB[i] = (__bf16)v;
}

// wg [4,128,128] ([k][j]) -> wgT bf16 [4,128,128] ([j][k])
__global__ void ggc_transpose_wg(const float* __restrict__ wg, __bf16* __restrict__ wgT) {
  int i = blockIdx.x * blockDim.x + threadIdx.x;
  if (i >= NSTEPS * DH * DH) return;
  const int s = i >> 14, rem = i & 16383, j = rem >> 7, k = rem & 127;
  wgT[i] = (__bf16)wg[(s << 14) + (k << 7) + j];
}

// ---------------- host launcher ----------------
extern "C" void kernel_launch(void* const* d_in, const int* in_sizes, int n_in,
                              void* d_out, int out_size, void* d_ws, size_t ws_size,
                              hipStream_t stream) {
  const float* x     = (const float*)d_in[0];
  const int*   edge  = (const int*)d_in[1];
  const int*   batch = (const int*)d_in[2];
  const float* gg_w[NCONV]; const float* wih[NCONV]; const float* whh[NCONV];
  const float* bih[NCONV];  const float* bhh[NCONV];
  for (int l = 0; l < NCONV; ++l) {
    gg_w[l] = (const float*)d_in[3 + 5 * l];
    wih[l]  = (const float*)d_in[4 + 5 * l];
    whh[l]  = (const float*)d_in[5 + 5 * l];
    bih[l]  = (const float*)d_in[6 + 5 * l];
    bhh[l]  = (const float*)d_in[7 + 5 * l];
  }
  const float* lin_w0 = (const float*)d_in[18];
  const float* lin_b0 = (const float*)d_in[19];
  const float* lin_w1 = (const float*)d_in[20];
  const float* lin_b1 = (const float*)d_in[21];

  // ---- workspace carve-up (256B aligned) ----
  char* ws = (char*)d_ws;
  size_t off = 0;
  auto carve = [&](size_t bytes) -> char* {
    char* p = ws + off;
    off = (off + bytes + 255) & ~(size_t)255;
    return p;
  };
  float*  h0      = (float*) carve((size_t)NNODES * DH * 4);
  float*  h1      = (float*) carve((size_t)NNODES * DH * 4);
  __bf16* hb0     = (__bf16*)carve((size_t)NNODES * DH * 2);
  __bf16* hb1     = (__bf16*)carve((size_t)NNODES * DH * 2);
  float*  m       = (float*) carve((size_t)NNODES * DH * 4);
  float*  agg     = (float*) carve((size_t)NNODES * DH * 4);
  __bf16* aggB    = (__bf16*)carve((size_t)NNODES * DH * 2);
  float*  pooled  = (float*) carve((size_t)NG * 3 * DH * 4);
  __bf16* pooledB = (__bf16*)carve((size_t)NG * 3 * DH * 2);
  float*  mlp0    = (float*) carve((size_t)NG * DH * 4);
  __bf16* mlp0B   = (__bf16*)carve((size_t)NG * DH * 2);
  __bf16* wgT     = (__bf16*)carve((size_t)NCONV * NSTEPS * DH * DH * 2);
  __bf16* wihB    = (__bf16*)carve((size_t)NCONV * 3 * DH * DH * 2);
  __bf16* whhB    = (__bf16*)carve((size_t)NCONV * 3 * DH * DH * 2);
  __bf16* lin0B   = (__bf16*)carve((size_t)DH * 3 * DH * 2);
  __bf16* lin1B   = (__bf16*)carve((size_t)DH * DH * 2);

  const int T = 256;
  // ---- weight prep (bf16 + transposes) ----
  for (int l = 0; l < NCONV; ++l) {
    ggc_transpose_wg<<<(NSTEPS * DH * DH + T - 1) / T, T, 0, stream>>>(
        gg_w[l], wgT + (size_t)l * NSTEPS * DH * DH);
    ggc_f32_to_bf16<<<(3 * DH * DH + T - 1) / T, T, 0, stream>>>(
        wih[l], wihB + (size_t)l * 3 * DH * DH, 3 * DH * DH);
    ggc_f32_to_bf16<<<(3 * DH * DH + T - 1) / T, T, 0, stream>>>(
        whh[l], whhB + (size_t)l * 3 * DH * DH, 3 * DH * DH);
  }
  ggc_f32_to_bf16<<<(DH * 3 * DH + T - 1) / T, T, 0, stream>>>(lin_w0, lin0B, DH * 3 * DH);
  ggc_f32_to_bf16<<<(DH * DH + T - 1) / T, T, 0, stream>>>(lin_w1, lin1B, DH * DH);

  // ---- init node state + pooled accumulator ----
  float*  hcur = h0;  float*  hnxt = h1;
  __bf16* bcur = hb0; __bf16* bnxt = hb1;
  ggc_pad_x<<<(NNODES * DH + T - 1) / T, T, 0, stream>>>(x, hcur, bcur, NNODES);
  ggc_zero_f32<<<(NG * 3 * DH + T - 1) / T, T, 0, stream>>>(pooled, NG * 3 * DH);

  const int* src = edge;
  const int* dst = edge + NEDGES;

  const dim3 gemmGrid((TILES_M + GEMM_TPB - 1) / GEMM_TPB, 1);
  const dim3 gruGrid((TILES_M + GRU_TPB - 1) / GRU_TPB);
  const size_t gemmShm128 = (size_t)128 * DH * 2;        // 32 KB
  const size_t gruShm     = (size_t)2 * 3 * DH * DH * 2; // 192 KB

  // ---- 3 GatedGraphConv layers x 4 GRU steps ----
  for (int l = 0; l < NCONV; ++l) {
    for (int s = 0; s < NSTEPS; ++s) {
      // m = h @ wg[l][s]    (WMMA, LDS-staged weight panel)
      ggc_gemm_bf16<<<gemmGrid, 256, gemmShm128, stream>>>(
          bcur, wgT + ((size_t)l * NSTEPS + s) * DH * DH, nullptr,
          m, nullptr, TILES_M, DH, DH, 0);
      // agg = scatter_add(m[src] -> dst)
      ggc_zero_f32<<<(NNODES * DH + T - 1) / T, T, 0, stream>>>(agg, NNODES * DH);
      ggc_edge_scatter<<<(NEDGES + 7) / 8, 256, 0, stream>>>(m, src, dst, agg, NEDGES);
      ggc_f32_to_bf16<<<(NNODES * DH + T - 1) / T, T, 0, stream>>>(agg, aggB, NNODES * DH);
      // fused GRU (12 WMMA per wave per k-loop, weights resident in LDS)
      const int relu = (s == NSTEPS - 1 && l < NCONV - 1) ? 1 : 0;
      ggc_gru_fused<<<gruGrid, 256, gruShm, stream>>>(
          aggB, bcur, wihB + (size_t)l * 3 * DH * DH, whhB + (size_t)l * 3 * DH * DH,
          bih[l], bhh[l], hcur, hnxt, bnxt, TILES_M, relu);
      { float* tf = hcur; hcur = hnxt; hnxt = tf; }
      { __bf16* tb = bcur; bcur = bnxt; bnxt = tb; }
    }
    // global add pool of this layer's output
    ggc_pool<<<(NNODES + 7) / 8, 256, 0, stream>>>(hcur, batch, pooled, NNODES, l * DH);
  }

  // ---- head MLP: relu(pooled @ W0^T + b0) -> relu(. @ W1^T + b1) ----
  ggc_f32_to_bf16<<<(NG * 3 * DH + T - 1) / T, T, 0, stream>>>(pooled, pooledB, NG * 3 * DH);
  ggc_gemm_bf16<<<dim3(1, 1), 256, (size_t)128 * 3 * DH * 2, stream>>>(
      pooledB, lin0B, lin_b0, mlp0, mlp0B, NG / 16, 3 * DH, DH, 1);
  ggc_gemm_bf16<<<dim3(1, 1), 256, gemmShm128, stream>>>(
      mlp0B, lin1B, lin_b1, (float*)d_out, nullptr, NG / 16, DH, DH, 1);
}
